// CRFLayer_81819126989373
// MI455X (gfx1250) — matrix operand order repaired and checked by reading, not compiled, not run on hardware
//
#include <hip/hip_runtime.h>

typedef _Float16 v16h __attribute__((ext_vector_type(16)));
typedef float    v8f  __attribute__((ext_vector_type(8)));

#define CRF_B 128
#define CRF_S 512
#define CRF_T 64
#define ROWS_PER_WG 16           // WMMA M dimension = batch rows per workgroup
#define NTHREADS 128             // 4 wave32 -> 4 j-tiles of 16
#define A_STRIDE 68              // f32 alpha row stride (272B, 16B aligned)
#define V_STRIDE 72              // f16 V row stride (144B, 16B aligned)

#define LN2F     0.69314718056f
#define INV_LN2F 1.44269504089f

__device__ __forceinline__ float fast_exp(float x) {
  return __builtin_amdgcn_exp2f(x * INV_LN2F);
}
__device__ __forceinline__ float fast_log(float x) {
  return __builtin_amdgcn_logf(x) * LN2F;
}

// ---------------------------------------------------------------------------
// Kernel 1: forward-algorithm normalizer.
//   grid = B/16 workgroups, 128 threads (4 wave32) each.
//   Per step: V[b,i] = exp(alpha[b,i] - m[b]) (f16, LDS, A-layout rows)
//             raw    = V @ exp(transitions)   (2x v_wmma_f32_16x16x32_f16)
//             alpha' = old + mkf*(log(raw)+m+em - old)   (branch-free fma)
// ---------------------------------------------------------------------------
__global__ __launch_bounds__(NTHREADS) void crf_normalizer_kernel(
    const float* __restrict__ emissions,        // [B,S,T]
    const float* __restrict__ transitions,      // [T,T]
    const float* __restrict__ startT,           // [T]
    const float* __restrict__ endT,             // [T]
    const unsigned char* __restrict__ mask,     // [B,S] (bool)
    float* __restrict__ den)                    // [B] out
{
  __shared__ float    sAlpha[ROWS_PER_WG][A_STRIDE];
  __shared__ _Float16 sV    [ROWS_PER_WG][V_STRIDE];
  __shared__ float    sM    [ROWS_PER_WG];
  __shared__ float    sMkF  [ROWS_PER_WG];     // mask as 0.0/1.0

  const int t      = threadIdx.x;
  const int wg     = blockIdx.x;
  const int wgbase = wg * ROWS_PER_WG;
  const int lane   = t & 31;
  const int wave   = t >> 5;
  const int jbase  = wave * 16;              // this wave's 16-column j-tile

  // element-wise phases: thread handles 8 contiguous cols of one row
  const int row8 = t >> 3;                   // 0..15
  const int sub8 = t & 7;                    // 0..7
  const int c0   = sub8 * 8;
  const int bg8  = wgbase + row8;            // global batch row

  // ---- preload constant B fragments: E[i][j] = exp(transitions[i][j]) ----
  // B 32x16 f16 layout: lane%16 = column N, lanes<16 hold K {0..7,16..23},
  // lanes>=16 hold K {8..15,24..31} (2 halves per VGPR).
  const int nb   = lane & 15;
  const int kb   = (lane < 16) ? 0 : 8;
  const int jcol = jbase + nb;
  v16h fB0, fB1;
  #pragma unroll
  for (int h = 0; h < 16; ++h) {
    fB0[h] = (_Float16)fast_exp(transitions[(kb + (h & 7) + ((h >> 3) << 4))      * CRF_T + jcol]);
    fB1[h] = (_Float16)fast_exp(transitions[(32 + kb + (h & 7) + ((h >> 3) << 4)) * CRF_T + jcol]);
  }

  // ---- alpha0 = startT + emissions[:,0,:] ----
  #pragma unroll
  for (int c = 0; c < 8; ++c) {
    int j = c0 + c;
    sAlpha[row8][j] = startT[j] + emissions[(size_t)bg8 * CRF_S * CRF_T + j];
  }
  __syncthreads();

  // A 16x32 f16 layout constants: lane%16 = row M; lanes<16 hold K {0..7,16..23},
  // lanes>=16 hold K {8..15,24..31} (contiguous 8-half groups -> ds_load_b128).
  const int rowA  = lane & 15;
  const int koff  = (lane < 16) ? 0 : 8;
  const int rbase = (lane < 16) ? 0 : 8;     // D-matrix M offset for this lane
  const int dcol  = jbase + (lane & 15);     // D-matrix column

  for (int s = 1; s < CRF_S; ++s) {
    // --- issue this step's emission loads early (latency overlap) ---
    float emv[8];
    #pragma unroll
    for (int r = 0; r < 8; ++r)
      emv[r] = emissions[((size_t)(wgbase + rbase + r) * CRF_S + s) * CRF_T + dcol];

    // --- row max of alpha: local max + 3-step shuffle reduce (no barrier) ---
    float av[8];
    #pragma unroll
    for (int c = 0; c < 8; ++c) av[c] = sAlpha[row8][c0 + c];
    float mm = av[0];
    #pragma unroll
    for (int c = 1; c < 8; ++c) mm = fmaxf(mm, av[c]);
    mm = fmaxf(mm, __shfl_xor(mm, 1));
    mm = fmaxf(mm, __shfl_xor(mm, 2));
    mm = fmaxf(mm, __shfl_xor(mm, 4));       // 8 row-threads all hold row max
    if (sub8 == 0) sM[row8] = mm;

    // --- V = exp(alpha - m) -> f16 LDS (one 16B row segment per thread) ---
    #pragma unroll
    for (int c = 0; c < 8; ++c)
      sV[row8][c0 + c] = (_Float16)fast_exp(av[c] - mm);

    // --- stage this step's mask (one float 0/1 per batch row) ---
    if (t < ROWS_PER_WG)
      sMkF[t] = mask[(size_t)(wgbase + t) * CRF_S + s] ? 1.0f : 0.0f;
    __syncthreads();

    // --- gather A fragments from LDS ---
    v16h a0, a1;
    const _Float16* vrow = &sV[rowA][0];
    #pragma unroll
    for (int h = 0; h < 8; ++h) {
      a0[h]     = vrow[koff + h];            // K =  koff..koff+7
      a0[8 + h] = vrow[16 + koff + h];       // K = 16+koff..
      a1[h]     = vrow[32 + koff + h];       // K = 32+koff..
      a1[8 + h] = vrow[48 + koff + h];       // K = 48+koff..
    }

    // --- old alpha + row data for branch-free masked update ---
    float oldv[8], sm8[8], mkf8[8];
    #pragma unroll
    for (int r = 0; r < 8; ++r) {
      oldv[r] = sAlpha[rbase + r][dcol];
      sm8[r]  = sM[rbase + r];
      mkf8[r] = sMkF[rbase + r];
    }

    // --- raw = V @ E  (16x16 tile, K=64 = two K=32 WMMAs, f32 accum) ---
    v8f acc = {};
    acc = __builtin_amdgcn_wmma_f32_16x16x32_f16(false, a0, false, fB0,
                                                 (short)0, acc, false, false);
    acc = __builtin_amdgcn_wmma_f32_16x16x32_f16(false, a1, false, fB1,
                                                 (short)0, acc, false, false);

    // --- alpha' = old + mkf*(log(raw)+m+em - old)  (raw >= exp(-0.1) > 0) ---
    #pragma unroll
    for (int r = 0; r < 8; ++r) {
      float na = fast_log(acc[r]) + sm8[r] + emv[r];
      sAlpha[rbase + r][dcol] = fmaf(mkf8[r], na - oldv[r], oldv[r]);
    }
    __syncthreads();
  }

  // ---- den[b] = logsumexp_j(alpha[b,j] + endT[j]) ----
  float xv[8];
  float fm = -3.4e38f;
  #pragma unroll
  for (int c = 0; c < 8; ++c) {
    xv[c] = sAlpha[row8][c0 + c] + endT[c0 + c];
    fm = fmaxf(fm, xv[c]);
  }
  fm = fmaxf(fm, __shfl_xor(fm, 1));
  fm = fmaxf(fm, __shfl_xor(fm, 2));
  fm = fmaxf(fm, __shfl_xor(fm, 4));         // row max across the 8 row-threads
  float ls = 0.f;
  #pragma unroll
  for (int c = 0; c < 8; ++c) ls += fast_exp(xv[c] - fm);
  ls += __shfl_xor(ls, 1);
  ls += __shfl_xor(ls, 2);
  ls += __shfl_xor(ls, 4);
  if (sub8 == 0) den[bg8] = fm + fast_log(ls);
}

// ---------------------------------------------------------------------------
// Kernel 2: numerator (gold-path score) + mean(den - num) -> scalar.
//   1 block, 128 threads, one thread per batch row.
// ---------------------------------------------------------------------------
__global__ __launch_bounds__(CRF_B) void crf_score_kernel(
    const float* __restrict__ emissions,
    const float* __restrict__ transitions,
    const float* __restrict__ startT,
    const float* __restrict__ endT,
    const int* __restrict__ tags,
    const unsigned char* __restrict__ mask,
    const float* __restrict__ den,
    float* __restrict__ out)
{
  const int b = threadIdx.x;
  const size_t tb = (size_t)b * CRF_S;

  int   t0    = tags[tb];
  float score = startT[t0] + emissions[(size_t)b * CRF_S * CRF_T + t0];
  int   cnt   = mask[tb] ? 1 : 0;
  int   prev  = t0;
  for (int s = 1; s < CRF_S; ++s) {
    int cur = tags[tb + s];
    float mp = mask[tb + s - 1] ? 1.f : 0.f;
    float mc = mask[tb + s]     ? 1.f : 0.f;
    float tr = transitions[prev * CRF_T + cur];
    float em = emissions[((size_t)b * CRF_S + s) * CRF_T + cur];
    score = fmaf(mp * mc, tr, score);
    score = fmaf(mc, em, score);
    cnt  += mask[tb + s] ? 1 : 0;
    prev = cur;
  }
  int lastTag = tags[tb + (cnt - 1)];
  score += endT[lastTag];

  float diff = den[b] - score;
  __shared__ float red[CRF_B];
  red[b] = diff;
  __syncthreads();
  for (int off = CRF_B / 2; off > 0; off >>= 1) {
    if (b < off) red[b] += red[b + off];
    __syncthreads();
  }
  if (b == 0) out[0] = red[0] * (1.0f / CRF_B);
}

// ---------------------------------------------------------------------------
extern "C" void kernel_launch(void* const* d_in, const int* in_sizes, int n_in,
                              void* d_out, int out_size, void* d_ws, size_t ws_size,
                              hipStream_t stream) {
  const float*         emissions   = (const float*)d_in[0];
  const float*         transitions = (const float*)d_in[1];
  const float*         startT      = (const float*)d_in[2];
  const float*         endT        = (const float*)d_in[3];
  const int*           tags        = (const int*)d_in[4];
  const unsigned char* maskp       = (const unsigned char*)d_in[5];

  float* den = (float*)d_ws;   // [B] denominators

  crf_normalizer_kernel<<<dim3(CRF_B / ROWS_PER_WG), dim3(NTHREADS), 0, stream>>>(
      emissions, transitions, startT, endT, maskp, den);
  crf_score_kernel<<<dim3(1), dim3(CRF_B), 0, stream>>>(
      emissions, transitions, startT, endT, tags, maskp, den, (float*)d_out);
}
